// FlaxHunyuanVideo15MidBlock_78941498900872
// MI455X (gfx1250) — compile-verified, matrix operand order checked by reading, not run on hardware
//
#include <hip/hip_runtime.h>

// ---------------------------------------------------------------------------
// HunyuanVideo15 MidBlock on gfx1250 (MI455X): resnet -> frame-causal attn ->
// resnet. All matmul work runs on v_wmma_f32_16x16x32_bf16 with register
// double-buffering sized to stay spill-free (16x64 wave tiles, ~140 VGPRs).
// ---------------------------------------------------------------------------

typedef __attribute__((ext_vector_type(16))) __bf16 v16bf;
typedef __attribute__((ext_vector_type(8)))  __bf16 v8bf;
typedef __attribute__((ext_vector_type(8)))  float  v8f;

#define CD     512
#define TD     8
#define HD     32
#define WD     32
#define NVOX   8192            // T*H*W
#define KCONV  13824           // 27*512
#define ATTN_SCALE 0.04419417382415922f  // 1/sqrt(512)
#define SQRT_C 22.627416997969522f       // sqrt(512)

__device__ __forceinline__ __bf16 f2bf(float f) {
  unsigned u = __float_as_uint(f);
  unsigned r = (u + 0x7FFFu + ((u >> 16) & 1u)) >> 16;
  unsigned short s = (unsigned short)r;
  __bf16 b;
  __builtin_memcpy(&b, &s, sizeof(b));
  return b;
}

// A/B fragment loader for v_wmma_f32_16x16x32_bf16.
// Caller passes  base + (row-or-col)*ld + kb  with kb = (lane>>4)*8.
// Halves 0..7 = K kb..kb+7, halves 8..15 = K kb+16..kb+23 (two 16B loads).
__device__ __forceinline__ v16bf load_frag(const __bf16* p) {
  v8bf lo = *(const v8bf*)(p);
  v8bf hi = *(const v8bf*)(p + 16);
  v16bf r;
#pragma unroll
  for (int j = 0; j < 8; ++j) { r[j] = lo[j]; r[8 + j] = hi[j]; }
  return r;
}

__device__ __forceinline__ v8f wmma_bf16(v16bf a, v16bf b, v8f c) {
  return __builtin_amdgcn_wmma_f32_16x16x32_bf16(false, a, false, b,
                                                 (short)0, c, false, false);
}

__device__ __forceinline__ v8f vzero8() {
  v8f z;
#pragma unroll
  for (int j = 0; j < 8; ++j) z[j] = 0.f;
  return z;
}

// ---------------------------------------------------------------------------
// RMSNorm (+ optional SiLU), f32 in -> bf16 out. One wave per C=512 row.
// blockDim = 256 (8 waves), grid = NVOX/8.
// ---------------------------------------------------------------------------
__global__ void k_rmsnorm(const float* __restrict__ x, const float* __restrict__ g,
                          __bf16* __restrict__ out, int do_silu) {
  int row  = blockIdx.x * (blockDim.x >> 5) + (threadIdx.x >> 5);
  int lane = threadIdx.x & 31;
  const float* xr = x + (size_t)row * CD;
  float v[16];
  float ss = 0.f;
#pragma unroll
  for (int i = 0; i < 16; ++i) { v[i] = xr[lane + i * 32]; ss += v[i] * v[i]; }
#pragma unroll
  for (int off = 16; off >= 1; off >>= 1) ss += __shfl_xor(ss, off, 32);
  float f = SQRT_C / (sqrtf(ss) + 1e-8f);
  __bf16* orow = out + (size_t)row * CD;
#pragma unroll
  for (int i = 0; i < 16; ++i) {
    int c = lane + i * 32;
    float y = v[i] * f * g[c];
    if (do_silu) y = y / (1.f + __expf(-y));
    orow[c] = f2bf(y);
  }
}

// w[27,Ci,Co] f32 -> wt[Co][27*Ci] bf16
__global__ void k_w3t(const float* __restrict__ w, __bf16* __restrict__ wt) {
  int o = blockIdx.x * blockDim.x + threadIdx.x;
  if (o >= 27 * CD * CD) return;
  int co = o / KCONV;
  int r  = o - co * KCONV;
  wt[o] = f2bf(w[(size_t)r * CD + co]);
}

// w[Ci,Co] f32 -> wt[Co][Ci] bf16
__global__ void k_w1t(const float* __restrict__ w, __bf16* __restrict__ wt) {
  int o = blockIdx.x * blockDim.x + threadIdx.x;
  if (o >= CD * CD) return;
  int co = o >> 9;
  int ci = o & 511;
  wt[o] = f2bf(w[(size_t)ci * CD + co]);
}

// ---------------------------------------------------------------------------
// Causal 3x3x3 conv as implicit GEMM; replicate pad == clamped indices.
// Wave = 16x64 tile. Outer kt(3) x kh(3) dynamic loops (clamps straight from
// loop indices, no div/mod); inner kw(3) x kc(16) region fully unrolled and
// register double-buffered so each WMMA step runs one load-step ahead.
// Block = 128 threads (4 waves); grid = (NVOX/64, CD/64).
// ---------------------------------------------------------------------------
__global__ void k_conv3d(const __bf16* __restrict__ act, const __bf16* __restrict__ wt,
                         const float* __restrict__ bias, const float* __restrict__ resid,
                         float* __restrict__ out) {
  int wv   = threadIdx.x >> 5;
  int lane = threadIdx.x & 31;
  int row0 = blockIdx.x * 64 + wv * 16;
  int col0 = blockIdx.y * 64;
  int m  = lane & 15;
  int kb = (lane >> 4) * 8;

  int vox = row0 + m;
  int t = vox >> 10, h = (vox >> 5) & 31, w = vox & 31;
  int wc0 = w - 1 < 0 ? 0 : w - 1;
  int wc1 = w;
  int wc2 = w + 1 > WD - 1 ? WD - 1 : w + 1;

  const __bf16* bpB = wt + (size_t)(col0 + m) * KCONV + kb;

  v8f acc[4];
#pragma unroll
  for (int i = 0; i < 4; ++i) acc[i] = vzero8();

  int bko = 0;                              // running tap*CD offset into Wt rows
  for (int kt = 0; kt < 3; ++kt) {
    int ts = t + kt - 2; ts = ts < 0 ? 0 : (ts > TD - 1 ? TD - 1 : ts);
    for (int kh = 0; kh < 3; ++kh) {
      int hs = h + kh - 1; hs = hs < 0 ? 0 : (hs > HD - 1 ? HD - 1 : hs);
      const __bf16* ab = act + (size_t)((ts * HD + hs) * WD) * CD + kb;
      const __bf16* arow[3] = { ab + (size_t)wc0 * CD,
                                ab + (size_t)wc1 * CD,
                                ab + (size_t)wc2 * CD };

      v16bf af[2], bf[2][4];
      af[0] = load_frag(arow[0]);
#pragma unroll
      for (int nt = 0; nt < 4; ++nt)
        bf[0][nt] = load_frag(bpB + bko + (size_t)nt * 16 * KCONV);

#pragma unroll
      for (int r = 0; r < 48; ++r) {        // r = kw*16 + kc  (all static)
        int cur = r & 1, nxt = cur ^ 1;
        if (r < 47) {
          int rn  = r + 1;
          int kwn = rn >> 4, kcn = rn & 15;
          af[nxt] = load_frag(arow[kwn] + kcn * 32);
#pragma unroll
          for (int nt = 0; nt < 4; ++nt)
            bf[nxt][nt] = load_frag(bpB + bko + kwn * CD + kcn * 32
                                        + (size_t)nt * 16 * KCONV);
        }
#pragma unroll
        for (int nt = 0; nt < 4; ++nt)
          acc[nt] = wmma_bf16(af[cur], bf[cur][nt], acc[nt]);
      }
      bko += 3 * CD;
    }
  }

  int g16 = lane >> 4;
#pragma unroll
  for (int nt = 0; nt < 4; ++nt) {
    int col = col0 + nt * 16 + (lane & 15);
    float bv = bias[col];
#pragma unroll
    for (int j = 0; j < 8; ++j) {
      size_t idx = (size_t)(row0 + j + 8 * g16) * CD + col;
      float r = acc[nt][j] + bv;
      if (resid) r += resid[idx];
      out[idx] = r;
    }
  }
}

// ---------------------------------------------------------------------------
// Dense GEMM: out[8192,512] = A(bf16 MxK) * Wt(bf16 NxK) + b.
// Wave = 16x64 tile, double-buffered k-loop. Output f32 (+resid) or bf16
// (optionally transposed for V^T). Block = 128; grid = (NVOX/64, CD/64).
// ---------------------------------------------------------------------------
__global__ void k_gemm512(const __bf16* __restrict__ A, const __bf16* __restrict__ Wt,
                          const float* __restrict__ bias, const float* __restrict__ resid,
                          float* __restrict__ outf, __bf16* __restrict__ outb, int otrans) {
  int wv   = threadIdx.x >> 5;
  int lane = threadIdx.x & 31;
  int row0 = blockIdx.x * 64 + wv * 16;
  int col0 = blockIdx.y * 64;
  int m  = lane & 15;
  int kb = (lane >> 4) * 8;

  const __bf16* ap  = A + (size_t)(row0 + m) * CD + kb;
  const __bf16* bp0 = Wt + (size_t)(col0 + m) * CD + kb;

  v8f acc[4];
#pragma unroll
  for (int i = 0; i < 4; ++i) acc[i] = vzero8();

  v16bf af[2], bf[2][4];
  af[0] = load_frag(ap);
#pragma unroll
  for (int nt = 0; nt < 4; ++nt) bf[0][nt] = load_frag(bp0 + (size_t)nt * 16 * CD);

#pragma unroll
  for (int kc = 0; kc < 16; ++kc) {
    int cur = kc & 1, nxt = cur ^ 1;
    if (kc < 15) {
      af[nxt] = load_frag(ap + (kc + 1) * 32);
#pragma unroll
      for (int nt = 0; nt < 4; ++nt)
        bf[nxt][nt] = load_frag(bp0 + (size_t)nt * 16 * CD + (kc + 1) * 32);
    }
#pragma unroll
    for (int nt = 0; nt < 4; ++nt)
      acc[nt] = wmma_bf16(af[cur], bf[cur][nt], acc[nt]);
  }

  int g16 = lane >> 4;
#pragma unroll
  for (int nt = 0; nt < 4; ++nt) {
    int col = col0 + nt * 16 + (lane & 15);
    float bv = bias[col];
#pragma unroll
    for (int j = 0; j < 8; ++j) {
      int row = row0 + j + 8 * g16;
      float r = acc[nt][j] + bv;
      if (outf) {
        size_t idx = (size_t)row * CD + col;
        if (resid) r += resid[idx];
        outf[idx] = r;
      } else if (otrans) {
        outb[(size_t)col * NVOX + row] = f2bf(r);   // V^T layout [C][S]
      } else {
        outb[(size_t)row * CD + col] = f2bf(r);
      }
    }
  }
}

// ---------------------------------------------------------------------------
// Attention pass 1: per-row softmax max & sum. One wave per 16-query tile,
// Q staged in LDS; 16-key tiles streamed; double-buffered score k-loop.
// Frame-causal mask == key loop bound. Block = 64 (2 waves), grid = NVOX/32.
// ---------------------------------------------------------------------------
__global__ void k_attn_stats(const __bf16* __restrict__ Q, const __bf16* __restrict__ K,
                             float* __restrict__ Mb, float* __restrict__ Lb) {
  __shared__ __bf16 qs[2][16 * CD];
  int wv   = threadIdx.x >> 5;
  int lane = threadIdx.x & 31;
  int row0 = blockIdx.x * 32 + wv * 16;

  {
    const uint4* src = (const uint4*)(Q + (size_t)row0 * CD);
    uint4* dst = (uint4*)qs[wv];
    for (int i = lane; i < 16 * CD / 8; i += 32) dst[i] = src[i];
  }

  int m  = lane & 15;
  int kb = (lane >> 4) * 8;
  const __bf16* qbase = qs[wv] + m * CD + kb;

  float mrow[8], lrow[8];
#pragma unroll
  for (int j = 0; j < 8; ++j) { mrow[j] = -3.0e38f; lrow[j] = 0.f; }

  int nkeys = ((row0 >> 10) + 1) << 10;
  for (int k0 = 0; k0 < nkeys; k0 += 16) {
    const __bf16* kbase = K + (size_t)(k0 + m) * CD + kb;
    v8f s = vzero8();
    v16bf qa[2], kf[2];
    qa[0] = load_frag(qbase);
    kf[0] = load_frag(kbase);
#pragma unroll
    for (int kc = 0; kc < 16; ++kc) {
      int cur = kc & 1, nxt = cur ^ 1;
      if (kc < 15) {
        qa[nxt] = load_frag(qbase + (kc + 1) * 32);
        kf[nxt] = load_frag(kbase + (kc + 1) * 32);
      }
      s = wmma_bf16(qa[cur], kf[cur], s);
    }
#pragma unroll
    for (int j = 0; j < 8; ++j) {
      float sv = s[j] * ATTN_SCALE;
      float rm = sv;
      rm = fmaxf(rm, __shfl_xor(rm, 1, 32));
      rm = fmaxf(rm, __shfl_xor(rm, 2, 32));
      rm = fmaxf(rm, __shfl_xor(rm, 4, 32));
      rm = fmaxf(rm, __shfl_xor(rm, 8, 32));
      float mn = fmaxf(mrow[j], rm);
      float e  = __expf(sv - mn);
      e += __shfl_xor(e, 1, 32);
      e += __shfl_xor(e, 2, 32);
      e += __shfl_xor(e, 4, 32);
      e += __shfl_xor(e, 8, 32);
      lrow[j] = lrow[j] * __expf(mrow[j] - mn) + e;
      mrow[j] = mn;
    }
  }
  if ((lane & 15) == 0) {
#pragma unroll
    for (int j = 0; j < 8; ++j) {
      int r = row0 + j + 8 * (lane >> 4);
      Mb[r] = mrow[j];
      Lb[r] = lrow[j];
    }
  }
}

// ---------------------------------------------------------------------------
// Attention pass 2: O = softmax(QK^T)*V with precomputed stats. Block owns
// one 16-query tile (4 waves x 128-col chunks). Per 32-key block: V frags are
// issued FIRST (consumed last -> overlap with 32 score WMMAs), score loops
// double-buffered, P re-swizzled D->A layout through per-wave LDS.
// grid = NVOX/16, block = 128.
// ---------------------------------------------------------------------------
__global__ void k_attn_out(const __bf16* __restrict__ Q, const __bf16* __restrict__ K,
                           const __bf16* __restrict__ Vt,
                           const float* __restrict__ Mb, const float* __restrict__ Lb,
                           __bf16* __restrict__ O) {
  __shared__ __bf16 qs[16 * CD];
  __shared__ __bf16 ps[4][16 * 32];
  int wv   = threadIdx.x >> 5;
  int lane = threadIdx.x & 31;
  int row0 = blockIdx.x * 16;

  {
    const uint4* src = (const uint4*)(Q + (size_t)row0 * CD);
    uint4* dst = (uint4*)qs;
    for (int i = threadIdx.x; i < 16 * CD / 8; i += 128) dst[i] = src[i];
  }
  __syncthreads();

  int m   = lane & 15;
  int kb  = (lane >> 4) * 8;
  int g16 = lane >> 4;
  int c0  = wv * 128;

  float mr[8], li[8];
#pragma unroll
  for (int j = 0; j < 8; ++j) {
    int r = row0 + j + 8 * g16;
    mr[j] = Mb[r];
    li[j] = 1.f / Lb[r];
  }

  v8f acc[8];
#pragma unroll
  for (int i = 0; i < 8; ++i) acc[i] = vzero8();

  const __bf16* qbase = qs + m * CD + kb;
  int nkeys = ((row0 >> 10) + 1) << 10;

  for (int k0 = 0; k0 < nkeys; k0 += 32) {
    // issue V fragment loads now; consumed after the score WMMAs
    v16bf vf[8];
#pragma unroll
    for (int nt = 0; nt < 8; ++nt)
      vf[nt] = load_frag(Vt + (size_t)(c0 + nt * 16 + m) * NVOX + k0 + kb);

#pragma unroll
    for (int half = 0; half < 2; ++half) {
      const __bf16* kbase = K + (size_t)(k0 + half * 16 + m) * CD + kb;
      v8f s = vzero8();
      v16bf qa[2], kf[2];
      qa[0] = load_frag(qbase);
      kf[0] = load_frag(kbase);
#pragma unroll
      for (int kc = 0; kc < 16; ++kc) {
        int cur = kc & 1, nxt = cur ^ 1;
        if (kc < 15) {
          qa[nxt] = load_frag(qbase + (kc + 1) * 32);
          kf[nxt] = load_frag(kbase + (kc + 1) * 32);
        }
        s = wmma_bf16(qa[cur], kf[cur], s);
      }
#pragma unroll
      for (int j = 0; j < 8; ++j) {
        float p = __expf(s[j] * ATTN_SCALE - mr[j]) * li[j];
        ps[wv][(j + 8 * g16) * 32 + half * 16 + (lane & 15)] = f2bf(p);
      }
    }
    // P(16x32) x V(32x128) from Vt[C][S]
    v16bf a = load_frag(ps[wv] + m * 32 + kb);
#pragma unroll
    for (int nt = 0; nt < 8; ++nt)
      acc[nt] = wmma_bf16(a, vf[nt], acc[nt]);
  }

#pragma unroll
  for (int nt = 0; nt < 8; ++nt) {
    int col = c0 + nt * 16 + (lane & 15);
#pragma unroll
    for (int j = 0; j < 8; ++j)
      O[(size_t)(row0 + j + 8 * g16) * CD + col] = f2bf(acc[nt][j]);
  }
}

// ---------------------------------------------------------------------------
// Launch sequence.  Workspace layout (bytes):
//   B0  bf16 act / attn-O            @ 0          (8,388,608)
//   B1  bf16 conv-Wt / (wtq..wtp,ML) @ 8,388,608  (14,155,776)
//   B2  f32 tmp                      @ 22,544,384 (16,777,216)
//   B3  f32 cur                      @ 39,321,600 (16,777,216)
//   Qb  bf16                         @ 56,098,816 (8,388,608)
//   Kb  bf16                         @ 64,487,424 (8,388,608)
//   Vt  bf16                         @ 72,876,032 (8,388,608)   total ~81.3MB
// ---------------------------------------------------------------------------
extern "C" void kernel_launch(void* const* d_in, const int* in_sizes, int n_in,
                              void* d_out, int out_size, void* d_ws, size_t ws_size,
                              hipStream_t stream) {
  const float* x     = (const float*)d_in[0];
  const float* r1_g1 = (const float*)d_in[1];
  const float* r1_w1 = (const float*)d_in[2];
  const float* r1_b1 = (const float*)d_in[3];
  const float* r1_g2 = (const float*)d_in[4];
  const float* r1_w2 = (const float*)d_in[5];
  const float* r1_b2 = (const float*)d_in[6];
  const float* at_g  = (const float*)d_in[7];
  const float* q_w   = (const float*)d_in[8];
  const float* q_b   = (const float*)d_in[9];
  const float* k_w   = (const float*)d_in[10];
  const float* k_b   = (const float*)d_in[11];
  const float* v_w   = (const float*)d_in[12];
  const float* v_b   = (const float*)d_in[13];
  const float* p_w   = (const float*)d_in[14];
  const float* p_b   = (const float*)d_in[15];
  const float* r2_g1 = (const float*)d_in[16];
  const float* r2_w1 = (const float*)d_in[17];
  const float* r2_b1 = (const float*)d_in[18];
  const float* r2_g2 = (const float*)d_in[19];
  const float* r2_w2 = (const float*)d_in[20];
  const float* r2_b2 = (const float*)d_in[21];

  char* ws = (char*)d_ws;
  __bf16* B0 = (__bf16*)(ws);
  __bf16* B1 = (__bf16*)(ws + 8388608);
  float*  B2 = (float*)(ws + 22544384);
  float*  B3 = (float*)(ws + 39321600);
  __bf16* Qb = (__bf16*)(ws + 56098816);
  __bf16* Kb = (__bf16*)(ws + 64487424);
  __bf16* Vt = (__bf16*)(ws + 72876032);
  __bf16* wtq = B1;
  __bf16* wtk = B1 + 262144;
  __bf16* wtv = B1 + 524288;
  __bf16* wtp = B1 + 786432;
  float*  Mb  = (float*)(ws + 8388608 + 2097152);
  float*  Lb  = Mb + NVOX;
  float*  out = (float*)d_out;

  dim3 gTile(NVOX / 64, CD / 64);   // 128 x 8 (wave = 16x64 tile)
  const int bTile = 128;
  const int gNorm = NVOX / 8, bNorm = 256;
  const int nW3 = 27 * CD * CD, gW3 = (nW3 + 255) / 256;
  const int nW1 = CD * CD,      gW1 = (nW1 + 255) / 256;

  // ---- resnet 1 ----
  k_rmsnorm<<<gNorm, bNorm, 0, stream>>>(x, r1_g1, B0, 1);
  k_w3t<<<gW3, 256, 0, stream>>>(r1_w1, B1);
  k_conv3d<<<gTile, bTile, 0, stream>>>(B0, B1, r1_b1, nullptr, B2);
  k_rmsnorm<<<gNorm, bNorm, 0, stream>>>(B2, r1_g2, B0, 1);
  k_w3t<<<gW3, 256, 0, stream>>>(r1_w2, B1);
  k_conv3d<<<gTile, bTile, 0, stream>>>(B0, B1, r1_b2, x, B3);   // cur = resnet1(x)

  // ---- attention ----
  k_rmsnorm<<<gNorm, bNorm, 0, stream>>>(B3, at_g, B0, 0);       // xn (no silu)
  k_w1t<<<gW1, 256, 0, stream>>>(q_w, wtq);
  k_w1t<<<gW1, 256, 0, stream>>>(k_w, wtk);
  k_w1t<<<gW1, 256, 0, stream>>>(v_w, wtv);
  k_w1t<<<gW1, 256, 0, stream>>>(p_w, wtp);
  k_gemm512<<<gTile, bTile, 0, stream>>>(B0, wtq, q_b, nullptr, nullptr, Qb, 0);
  k_gemm512<<<gTile, bTile, 0, stream>>>(B0, wtk, k_b, nullptr, nullptr, Kb, 0);
  k_gemm512<<<gTile, bTile, 0, stream>>>(B0, wtv, v_b, nullptr, nullptr, Vt, 1); // V^T
  k_attn_stats<<<NVOX / 32, 64, 0, stream>>>(Qb, Kb, Mb, Lb);
  k_attn_out<<<NVOX / 16, 128, 0, stream>>>(Qb, Kb, Vt, Mb, Lb, B0);             // O bf16
  k_gemm512<<<gTile, bTile, 0, stream>>>(B0, wtp, p_b, B3, B2, nullptr, 0);      // cur2

  // ---- resnet 2 ----
  k_rmsnorm<<<gNorm, bNorm, 0, stream>>>(B2, r2_g1, B0, 1);
  k_w3t<<<gW3, 256, 0, stream>>>(r2_w1, B1);
  k_conv3d<<<gTile, bTile, 0, stream>>>(B0, B1, r2_b1, nullptr, B3);
  k_rmsnorm<<<gNorm, bNorm, 0, stream>>>(B3, r2_g2, B0, 1);
  k_w3t<<<gW3, 256, 0, stream>>>(r2_w2, B1);
  k_conv3d<<<gTile, bTile, 0, stream>>>(B0, B1, r2_b2, B2, out);
}
